// IterativeLSTM_12197707120743
// MI455X (gfx1250) — matrix operand order repaired and checked by reading, not compiled, not run on hardware
//
#include <hip/hip_runtime.h>
#include <hip/hip_bf16.h>
#include <stdint.h>

typedef __bf16        v16bf __attribute__((ext_vector_type(16)));
typedef float         v8f   __attribute__((ext_vector_type(8)));
typedef unsigned int  u32x4 __attribute__((ext_vector_type(4)));
typedef int           i32x4 __attribute__((ext_vector_type(4)));

#define B_DIM 8192
#define I_DIM 1024
#define H_DIM 1024
#define KTOT  (I_DIM + H_DIM)        // unified K = 2048
#define BK    32
#define NK    (KTOT / BK)            // 64 chunks
#define LSTRIDE 40                   // LDS row stride (80 B: 16B-aligned, conflict-free)

// fast-path tile: 128 rows x (4 gates x 64 j-cols), 512 threads (16 waves)
#define BM  128
#define BNJ 64
#define TPB_WS 512

// CDNA5 async global->LDS (ASYNCcnt-tracked), guarded: fall back to reg staging
#if defined(__has_builtin)
#if __has_builtin(__builtin_amdgcn_global_load_async_to_lds_b128)
#define HAVE_ASYNC_LDS 1
#endif
#endif
#ifndef HAVE_ASYNC_LDS
#define HAVE_ASYNC_LDS 0
#endif

#if HAVE_ASYNC_LDS
// builtin signature (from clang diagnostic): (AS1 int4*, AS3 int4*, int, int)
#define GPTR(p) ((__attribute__((address_space(1))) i32x4*)(p))
#define LPTR(p) ((__attribute__((address_space(3))) i32x4*)(p))
#endif

// ---- fp32 -> packed 2x bf16 (RNE) ----------------------------------------
__device__ __forceinline__ unsigned pack_bf16x2(float a, float b) {
    unsigned ua = __float_as_uint(a), ub = __float_as_uint(b);
    ua += 0x7FFFu + ((ua >> 16) & 1u);
    ub += 0x7FFFu + ((ub >> 16) & 1u);
    return (ua >> 16) | (ub & 0xFFFF0000u);
}

// ---- branch-free activations (v_exp_f32 + v_rcp_f32, saturate at +-inf) --
__device__ __forceinline__ float sigmoid_fast(float x) {
    const float e = __builtin_amdgcn_exp2f(-1.44269504088896340736f * x);
    return __builtin_amdgcn_rcpf(1.0f + e);
}
__device__ __forceinline__ float tanh_fast(float x) {
    const float e = __builtin_amdgcn_exp2f(2.88539008177792681472f * x);
    return 1.0f - 2.0f * __builtin_amdgcn_rcpf(1.0f + e);
}

union Fr { v16bf v; u32x4 q2[2]; };

// ===========================================================================
// Pre-pass: concat-convert two fp32 [rows][1024] sources into bf16 [rows][2048]
// ===========================================================================
__global__ __launch_bounds__(256)
void cvt_concat_kernel(const float* __restrict__ lo, const float* __restrict__ hi,
                       unsigned short* __restrict__ dst, int rows) {
    const size_t tid8 = ((size_t)blockIdx.x * blockDim.x + threadIdx.x) * 8;
    if (tid8 >= (size_t)rows * KTOT) return;
    const int row = (int)(tid8 / KTOT);
    const int col = (int)(tid8 % KTOT);
    const float* src = (col < I_DIM) ? (lo + (size_t)row * I_DIM + col)
                                     : (hi + (size_t)row * I_DIM + (col - I_DIM));
    const float4* s4 = (const float4*)src;
    const float4 f0 = s4[0], f1 = s4[1];
    u32x4 p;
    p[0] = pack_bf16x2(f0.x, f0.y); p[1] = pack_bf16x2(f0.z, f0.w);
    p[2] = pack_bf16x2(f1.x, f1.y); p[3] = pack_bf16x2(f1.z, f1.w);
    *(u32x4*)(dst + tid8) = p;
}

// ===========================================================================
// Fast path: bf16 from workspace, 128x(4x64) tile, 16 waves, acc[4][2]/wave
// ===========================================================================
__global__ __launch_bounds__(TPB_WS)
void lstm_wmma_ws(const unsigned short* __restrict__ Abf,   // [8192][2048]
                  const unsigned short* __restrict__ Wbf,   // [4096][2048]
                  const float* __restrict__ c_prev,
                  const float* __restrict__ bx,
                  const float* __restrict__ bh,
                  float* __restrict__ out_c,
                  float* __restrict__ out_h) {
    __shared__ __align__(16) unsigned short sA[2][BM * LSTRIDE];       // [m][k]
    __shared__ __align__(16) unsigned short sB[2][4 * BNJ * LSTRIDE];  // [q*64+j][k]

    const int tid    = threadIdx.x;
    const int lane   = tid & 31;
    const int wave   = tid >> 5;
    const int wave_m = wave & 3;    // 4 waves along M: 32 rows (2 m-tiles) each
    const int wave_n = wave >> 2;   // 4 waves along j: 16 cols each

    const int m_block = blockIdx.x * BM;
    const int h0      = blockIdx.y * BNJ;
    const int j0      = wave_n * 16;
    const int ncol    = lane & 15;
    const int mhalf   = (lane >= 16) ? 8 : 0;

    // loaders (512 threads): A = 16 B (1x b128) per thread; B = 32 B (2x b128)
    const int rowA = tid >> 2, quarterA = (tid & 3) * 8;   // 128 rows x 4 quarters
    const int rowB = tid >> 1, halfB    = (tid & 1) * 16;  // 256 rows x 2 halves
    const int qB   = rowB >> 6, jB = rowB & 63;
    const unsigned short* gA = Abf + (size_t)(m_block + rowA) * KTOT + quarterA;
    const unsigned short* gW = Wbf + (size_t)(qB * 1024 + h0 + jB) * KTOT + halfB;

    // accumulators; bias folded into init (depends only on lane/N column)
    v8f acc[4][2];
    #pragma unroll
    for (int q = 0; q < 4; ++q) {
        const int g = q * 1024 + h0 + j0 + ncol;
        const float bias = bx[g] + bh[g];
        v8f bv;
        #pragma unroll
        for (int r = 0; r < 8; ++r) bv[r] = bias;
        acc[q][0] = bv; acc[q][1] = bv;
    }

    // LDS fragment addressing (ISA bf16 A 16x32 / B 32x16 lane layouts)
    const int arow0  = wave_m * 32 + (lane & 15);
    const int a_koff = (lane >= 16) ? 8 : 0;
    const int b_koff = (lane >= 16) ? 8 : 0;

#if HAVE_ASYNC_LDS
    auto issue_async = [&](int kc, int buf) {
        const unsigned short* ga = gA + (size_t)kc * BK;
        const unsigned short* gw = gW + (size_t)kc * BK;
        unsigned short* la = &sA[buf][rowA * LSTRIDE + quarterA];
        unsigned short* lb = &sB[buf][rowB * LSTRIDE + halfB];
        __builtin_amdgcn_global_load_async_to_lds_b128(GPTR(ga),     LPTR(la),     0, 0);
        __builtin_amdgcn_global_load_async_to_lds_b128(GPTR(gw),     LPTR(lb),     0, 0);
        __builtin_amdgcn_global_load_async_to_lds_b128(GPTR(gw + 8), LPTR(lb + 8), 0, 0);
    };
    issue_async(0, 0);
#else
    u32x4 ra, rb[2];
    auto prefetch = [&](int kc) {
        ra = *(const u32x4*)(gA + (size_t)kc * BK);
        const u32x4* pb = (const u32x4*)(gW + (size_t)kc * BK);
        rb[0] = pb[0]; rb[1] = pb[1];
    };
    auto lds_store = [&](int buf) {
        *(u32x4*)&sA[buf][rowA * LSTRIDE + quarterA] = ra;
        u32x4* db = (u32x4*)&sB[buf][rowB * LSTRIDE + halfB];
        db[0] = rb[0]; db[1] = rb[1];
    };
    prefetch(0);
    lds_store(0);
#endif

    for (int kc = 0; kc < NK; ++kc) {
#if HAVE_ASYNC_LDS
        asm volatile("s_wait_asynccnt 0x0" ::: "memory");  // my LDS fills done
#endif
        __syncthreads();                                   // all waves' fills visible
        const int buf = kc & 1;

#if HAVE_ASYNC_LDS
        if (kc + 1 < NK) issue_async(kc + 1, buf ^ 1);     // fire-and-forget next chunk
#else
        if (kc + 1 < NK) prefetch(kc + 1);
#endif

        Fr afr[2];
        #pragma unroll
        for (int mt = 0; mt < 2; ++mt) {
            const unsigned short* p = &sA[buf][(arow0 + mt * 16) * LSTRIDE + a_koff];
            afr[mt].q2[0] = *(const u32x4*)p;
            afr[mt].q2[1] = *(const u32x4*)(p + 16);
        }
        #pragma unroll
        for (int q = 0; q < 4; ++q) {
            Fr bfr;
            const unsigned short* p = &sB[buf][(q * BNJ + j0 + ncol) * LSTRIDE + b_koff];
            bfr.q2[0] = *(const u32x4*)p;
            bfr.q2[1] = *(const u32x4*)(p + 16);
            #pragma unroll
            for (int mt = 0; mt < 2; ++mt)
                acc[q][mt] = __builtin_amdgcn_wmma_f32_16x16x32_bf16(
                    false, afr[mt].v, false, bfr.v,
                    (short)0, acc[q][mt], false, false);
        }

#if !HAVE_ASYNC_LDS
        if (kc + 1 < NK) lds_store(buf ^ 1);
#endif
    }

    // fused epilogue: i/f/o/g for one (b,h) share lane & vgpr slot
    const int colg = h0 + j0 + ncol;
    const int mrow = m_block + wave_m * 32 + mhalf;
    #pragma unroll
    for (int mt = 0; mt < 2; ++mt) {
        #pragma unroll
        for (int r = 0; r < 8; ++r) {
            const size_t idx = (size_t)(mrow + mt * 16 + r) * (size_t)H_DIM + colg;
            const float ig = sigmoid_fast(acc[0][mt][r]);
            const float fg = sigmoid_fast(acc[1][mt][r]);
            const float og = sigmoid_fast(acc[2][mt][r]);
            const float gg = tanh_fast(acc[3][mt][r]);
            const float c  = fg * c_prev[idx] + ig * gg;
            out_c[idx] = c;
            out_h[idx] = og * tanh_fast(c);
        }
    }
}

// ===========================================================================
// Fallback (ws too small): fused-conversion kernel
// ===========================================================================
__global__ __launch_bounds__(256)
void lstm_wmma_fused(const float* __restrict__ x,
                     const float* __restrict__ c_prev,
                     const float* __restrict__ h_prev,
                     const float* __restrict__ Wx,
                     const float* __restrict__ bx,
                     const float* __restrict__ Wh,
                     const float* __restrict__ bh,
                     float* __restrict__ out_c,
                     float* __restrict__ out_h) {
    __shared__ __align__(16) unsigned short sA[2][128 * LSTRIDE];
    __shared__ __align__(16) unsigned short sB[2][128 * LSTRIDE];

    const int tid    = threadIdx.x;
    const int lane   = tid & 31;
    const int wave   = tid >> 5;
    const int wave_m = wave & 3;
    const int wave_n = wave >> 2;

    const int m_block = blockIdx.x * 128;
    const int h0      = blockIdx.y * 32;
    const int j0      = wave_n * 16;
    const int ncol    = lane & 15;
    const int mhalf   = (lane >= 16) ? 8 : 0;

    const int lrow  = tid >> 1;
    const int lhalf = (tid & 1) * 16;
    const int g_ld  = (lrow >> 5) * 1024 + h0 + (lrow & 31);

    v8f acc[4][2];
    #pragma unroll
    for (int q = 0; q < 4; ++q) {
        const int g = q * 1024 + h0 + j0 + ncol;
        const float bias = bx[g] + bh[g];
        v8f bv;
        #pragma unroll
        for (int r = 0; r < 8; ++r) bv[r] = bias;
        acc[q][0] = bv; acc[q][1] = bv;
    }

    const int arow0  = wave_m * 32 + (lane & 15);
    const int a_koff = (lane >= 16) ? 8 : 0;
    const int b_koff = (lane >= 16) ? 8 : 0;

    float4 pa[4], pb[4];
    auto prefetch = [&](int kc) {
        const int k0 = kc * BK;
        const float* Abase; const float* Wbase; int kk;
        if (k0 < I_DIM) { Abase = x;      Wbase = Wx; kk = k0; }
        else            { Abase = h_prev; Wbase = Wh; kk = k0 - I_DIM; }
        const float4* sa = (const float4*)(Abase + (size_t)(m_block + lrow) * I_DIM + kk + lhalf);
        const float4* sb = (const float4*)(Wbase + (size_t)g_ld * I_DIM + kk + lhalf);
        #pragma unroll
        for (int i = 0; i < 4; ++i) { pa[i] = sa[i]; pb[i] = sb[i]; }
    };
    auto lds_store = [&](int buf) {
        u32x4 qa0, qa1, qb0, qb1;
        qa0[0] = pack_bf16x2(pa[0].x, pa[0].y); qa0[1] = pack_bf16x2(pa[0].z, pa[0].w);
        qa0[2] = pack_bf16x2(pa[1].x, pa[1].y); qa0[3] = pack_bf16x2(pa[1].z, pa[1].w);
        qa1[0] = pack_bf16x2(pa[2].x, pa[2].y); qa1[1] = pack_bf16x2(pa[2].z, pa[2].w);
        qa1[2] = pack_bf16x2(pa[3].x, pa[3].y); qa1[3] = pack_bf16x2(pa[3].z, pa[3].w);
        qb0[0] = pack_bf16x2(pb[0].x, pb[0].y); qb0[1] = pack_bf16x2(pb[0].z, pb[0].w);
        qb0[2] = pack_bf16x2(pb[1].x, pb[1].y); qb0[3] = pack_bf16x2(pb[1].z, pb[1].w);
        qb1[0] = pack_bf16x2(pb[2].x, pb[2].y); qb1[1] = pack_bf16x2(pb[2].z, pb[2].w);
        qb1[2] = pack_bf16x2(pb[3].x, pb[3].y); qb1[3] = pack_bf16x2(pb[3].z, pb[3].w);
        u32x4* da = (u32x4*)&sA[buf][lrow * LSTRIDE + lhalf];
        u32x4* db = (u32x4*)&sB[buf][lrow * LSTRIDE + lhalf];
        da[0] = qa0; da[1] = qa1; db[0] = qb0; db[1] = qb1;
    };

    prefetch(0);
    lds_store(0);

    for (int kc = 0; kc < NK; ++kc) {
        __syncthreads();
        const int buf = kc & 1;
        if (kc + 1 < NK) prefetch(kc + 1);

        Fr afr[2];
        #pragma unroll
        for (int mt = 0; mt < 2; ++mt) {
            const unsigned short* p = &sA[buf][(arow0 + mt * 16) * LSTRIDE + a_koff];
            afr[mt].q2[0] = *(const u32x4*)p;
            afr[mt].q2[1] = *(const u32x4*)(p + 16);
        }
        #pragma unroll
        for (int q = 0; q < 4; ++q) {
            Fr bfr;
            const unsigned short* p = &sB[buf][(q * 32 + j0 + ncol) * LSTRIDE + b_koff];
            bfr.q2[0] = *(const u32x4*)p;
            bfr.q2[1] = *(const u32x4*)(p + 16);
            #pragma unroll
            for (int mt = 0; mt < 2; ++mt)
                acc[q][mt] = __builtin_amdgcn_wmma_f32_16x16x32_bf16(
                    false, afr[mt].v, false, bfr.v,
                    (short)0, acc[q][mt], false, false);
        }

        if (kc + 1 < NK) lds_store(buf ^ 1);
    }

    const int colg = h0 + j0 + ncol;
    const int mrow = m_block + wave_m * 32 + mhalf;
    #pragma unroll
    for (int mt = 0; mt < 2; ++mt) {
        #pragma unroll
        for (int r = 0; r < 8; ++r) {
            const size_t idx = (size_t)(mrow + mt * 16 + r) * (size_t)H_DIM + colg;
            const float ig = sigmoid_fast(acc[0][mt][r]);
            const float fg = sigmoid_fast(acc[1][mt][r]);
            const float og = sigmoid_fast(acc[2][mt][r]);
            const float gg = tanh_fast(acc[3][mt][r]);
            const float c  = fg * c_prev[idx] + ig * gg;
            out_c[idx] = c;
            out_h[idx] = og * tanh_fast(c);
        }
    }
}

extern "C" void kernel_launch(void* const* d_in, const int* in_sizes, int n_in,
                              void* d_out, int out_size, void* d_ws, size_t ws_size,
                              hipStream_t stream) {
    const float* x     = (const float*)d_in[0];
    const float* cprev = (const float*)d_in[1];
    const float* hprev = (const float*)d_in[2];
    const float* Wx    = (const float*)d_in[3];
    const float* bx    = (const float*)d_in[4];
    const float* Wh    = (const float*)d_in[5];
    const float* bh    = (const float*)d_in[6];
    float* out_c = (float*)d_out;
    float* out_h = out_c + (size_t)B_DIM * (size_t)H_DIM;

    const size_t nA   = (size_t)B_DIM * KTOT;        // bf16 [x|h]
    const size_t nW   = (size_t)(4 * H_DIM) * KTOT;  // bf16 [Wx|Wh]
    const size_t need = (nA + nW) * sizeof(unsigned short);   // ~50.3 MB

    if (ws_size >= need) {
        unsigned short* Abf = (unsigned short*)d_ws;
        unsigned short* Wbf = Abf + nA;
        cvt_concat_kernel<<<(unsigned)(nA / 8 / 256), 256, 0, stream>>>(x, hprev, Abf, B_DIM);
        cvt_concat_kernel<<<(unsigned)(nW / 8 / 256), 256, 0, stream>>>(Wx, Wh, Wbf, 4 * H_DIM);
        dim3 grid(B_DIM / BM, H_DIM / BNJ);          // 64 x 16 = 1024 blocks
        lstm_wmma_ws<<<grid, TPB_WS, 0, stream>>>(Abf, Wbf, cprev, bx, bh, out_c, out_h);
    } else {
        dim3 grid(B_DIM / 128, H_DIM / 32);          // 64 x 32 = 2048 blocks
        lstm_wmma_fused<<<grid, 256, 0, stream>>>(x, cprev, hprev, Wx, bx, Wh, bh,
                                                  out_c, out_h);
    }
}
